// ProteinFeaturesMembrane_42176578846968
// MI455X (gfx1250) — matrix operand order repaired and checked by reading, not compiled, not run on hardware
//
#include <hip/hip_runtime.h>

// ---------------- problem constants ----------------
#define BB 4
#define NN 1024
#define TOPK 48
#define NUM_RBF 16
#define MAX_REL 32
#define EDGE_IN 416      // 16 positional + 25 groups * 16 rbf
#define EDGE_F 128
#define NODE_F 128
#define NEDGE (BB * NN * TOPK)       // 196608
#define FSTRIDE 416                  // bf16 elements per feature row (mult of 8)
#define WSTRIDE 416                  // bf16 elements per weight column (mult of 8)
#define TILES_PER_BLOCK 16           // 16 edges per M-tile, 16 tiles/block
#define KCHUNKS 13                   // 416 / 32

typedef __attribute__((ext_vector_type(16))) __bf16 v16bf;
typedef __attribute__((ext_vector_type(8)))  __bf16 v8bf;
typedef __attribute__((ext_vector_type(8)))  float  v8f;

// ---------------- kernel 1: derived coordinates (N,Ca,C,O,Cb) ----------------
__global__ void coords_kernel(const float* __restrict__ X,
                              float* __restrict__ coords, int total) {
  int idx = blockIdx.x * blockDim.x + threadIdx.x;
  if (idx >= total) return;
  const float* p = X + (size_t)idx * 12;
  float Nx = p[0],  Ny = p[1],  Nz = p[2];
  float Cax = p[3], Cay = p[4], Caz = p[5];
  float Cx = p[6],  Cy = p[7],  Cz = p[8];
  float Ox = p[9],  Oy = p[10], Oz = p[11];
  float bx = Cax - Nx,  by = Cay - Ny,  bz = Caz - Nz;
  float cx = Cx - Cax,  cy = Cy - Cay,  cz = Cz - Caz;
  float ax = by * cz - bz * cy;
  float ay = bz * cx - bx * cz;
  float az = bx * cy - by * cx;
  float Cbx = -0.58273431f * ax + 0.56802827f * bx - 0.54067466f * cx + Cax;
  float Cby = -0.58273431f * ay + 0.56802827f * by - 0.54067466f * cy + Cay;
  float Cbz = -0.58273431f * az + 0.56802827f * bz - 0.54067466f * cz + Caz;
  float* o = coords + (size_t)idx * 16;
  o[0]  = Nx;  o[1]  = Ny;  o[2]  = Nz;
  o[3]  = Cax; o[4]  = Cay; o[5]  = Caz;
  o[6]  = Cx;  o[7]  = Cy;  o[8]  = Cz;
  o[9]  = Ox;  o[10] = Oy;  o[11] = Oz;
  o[12] = Cbx; o[13] = Cby; o[14] = Cbz;
  o[15] = 0.f;
}

// ---------------- kernel 2: node embedding V = onehot @ node_w + node_b ------
__global__ void node_kernel(const int* __restrict__ lbl,
                            const float* __restrict__ node_w,
                            const float* __restrict__ node_b,
                            float* __restrict__ outV) {
  int i = blockIdx.x;
  int f = threadIdx.x;
  int l = lbl[i];
  outV[(size_t)i * NODE_F + f] = node_w[l * NODE_F + f] + node_b[f];
}

// ---------------- kernel 3: masked distances + top-k (lax.top_k semantics) ---
__device__ __forceinline__ unsigned long long make_key(float v, int j) {
  if (v < 0.f) return 0ull;
  unsigned int fb = __float_as_uint(v);  // monotonic for v >= 0
  return ((unsigned long long)fb << 32) | (unsigned int)(~j);
}

__global__ void topk_kernel(const float* __restrict__ coords,
                            const float* __restrict__ mask,
                            int* __restrict__ eidx_ws,
                            float* __restrict__ out_eidx) {
  __shared__ float Dv[NN];
  __shared__ float redf[256];
  __shared__ unsigned long long red[256];

  int row = blockIdx.x;            // b*N + i
  int b = row >> 10;
  int i = row & (NN - 1);
  int tid = threadIdx.x;

  const float* cbase = coords + ((size_t)b << 10) * 16;
  float xi = cbase[i * 16 + 3], yi = cbase[i * 16 + 4], zi = cbase[i * 16 + 5];
  float mi = mask[row];

  float lmax = 0.f;
  for (int q = 0; q < 4; ++q) {
    int j = tid + q * 256;
    float dx = xi - cbase[j * 16 + 3];
    float dy = yi - cbase[j * 16 + 4];
    float dz = zi - cbase[j * 16 + 5];
    float d = sqrtf(dx * dx + dy * dy + dz * dz + 1e-6f);
    float m2 = mi * mask[(b << 10) + j];
    float dm = m2 * d;
    Dv[j] = dm;
    lmax = fmaxf(lmax, dm);
  }
  redf[tid] = lmax;
  __syncthreads();
  for (int s = 128; s > 0; s >>= 1) {
    if (tid < s) redf[tid] = fmaxf(redf[tid], redf[tid + s]);
    __syncthreads();
  }
  float rmax = redf[0];
  __syncthreads();
  for (int q = 0; q < 4; ++q) {
    int j = tid + q * 256;
    float m2 = mi * mask[(b << 10) + j];
    Dv[j] = Dv[j] + (1.f - m2) * rmax;   // D_adjust
  }
  __syncthreads();

  for (int k = 0; k < TOPK; ++k) {
    unsigned long long best = 0ull;
    for (int q = 0; q < 4; ++q) {
      int j = tid + q * 256;
      unsigned long long kk = make_key(Dv[j], j);
      best = (kk > best) ? kk : best;
    }
    red[tid] = best;
    __syncthreads();
    for (int s = 128; s > 0; s >>= 1) {
      if (tid < s) {
        unsigned long long o = red[tid + s];
        if (o > red[tid]) red[tid] = o;
      }
      __syncthreads();
    }
    if (tid == 0) {
      unsigned long long kk = red[0];
      int j = (int)(~(unsigned int)(kk & 0xFFFFFFFFull));
      eidx_ws[(size_t)row * TOPK + k] = j;
      out_eidx[(size_t)row * TOPK + k] = (float)j;
      Dv[j] = -1.f;                       // remove from future rounds
    }
    __syncthreads();
  }
}

// ---------------- kernel 4: fused edge features + WMMA GEMM ------------------
// atom ids: 0=N 1=Ca 2=C 3=O 4=Cb ; group 0 = Ca-Ca (D_neighbors), then 24 pairs
__constant__ int GA[25] = {1, 0,2,3,4, 1,1,1,1, 0,0,0, 4,4, 3, 0,2,3,4, 2,3,4, 2,3, 2};
__constant__ int GB[25] = {1, 0,2,3,4, 0,2,3,4, 2,3,4, 2,3, 2, 1,1,1,1, 0,0,0, 4,4, 3};

__launch_bounds__(256)
__global__ void edge_kernel(const float* __restrict__ coords,
                            const int* __restrict__ eidx,
                            const int* __restrict__ Ridx,
                            const int* __restrict__ chain,
                            const float* __restrict__ pos_w,
                            const float* __restrict__ pos_b,
                            const float* __restrict__ edge_w,
                            const float* __restrict__ edge_b,
                            float* __restrict__ outE) {
  extern __shared__ char smem[];
  __bf16* wT     = (__bf16*)smem;                 // [128][WSTRIDE] transposed weights
  __bf16* feat   = wT + EDGE_F * WSTRIDE;         // [16][FSTRIDE] feature tile
  float*  s_posw = (float*)(feat + 16 * FSTRIDE); // 66*16
  float*  s_posb = s_posw + 66 * 16;              // 16
  float*  s_edgeb = s_posb + 16;                  // 128

  const int tid  = threadIdx.x;
  const int lane = tid & 31;
  const int wave = tid >> 5;                      // 0..7, owns cols wave*16..+15

  // ---- stage weights (f32 -> bf16, transposed) and small tables into LDS ----
  for (int idx = tid; idx < EDGE_IN * EDGE_F; idx += 256) {
    int k = idx >> 7;          // row of edge_w (feature)
    int n = idx & 127;         // col (output feature)
    wT[n * WSTRIDE + k] = (__bf16)edge_w[idx];
  }
  for (int idx = tid; idx < 66 * 16; idx += 256) s_posw[idx] = pos_w[idx];
  if (tid < 16)  s_posb[tid]  = pos_b[tid];
  if (tid < 128) s_edgeb[tid] = edge_b[tid];
  __syncthreads();

  for (int ti = 0; ti < TILES_PER_BLOCK; ++ti) {
    const int tileBase = (blockIdx.x * TILES_PER_BLOCK + ti) * 16;  // first edge of tile
    __syncthreads();   // previous tile's WMMA reads of feat are done

    // ---- build 16 x 416 bf16 feature tile: thread = (edge e, group g) ------
    {
      int e = tid & 15;
      int g = tid >> 4;                       // 0..15
      int edge = tileBase + e;
      int b   = edge / (NN * TOPK);
      int rem = edge - b * (NN * TOPK);
      int i   = rem / TOPK;
      int j   = eidx[edge];
      const float* ci = coords + (((size_t)b << 10) + i) * 16;
      const float* cj = coords + (((size_t)b << 10) + j) * 16;

      for (int gg = g; gg < 25; gg += 16) {
        int a0 = GA[gg] * 3, b0 = GB[gg] * 3;
        float dx = ci[a0 + 0] - cj[b0 + 0];
        float dy = ci[a0 + 1] - cj[b0 + 1];
        float dz = ci[a0 + 2] - cj[b0 + 2];
        float D = sqrtf(dx * dx + dy * dy + dz * dz + 1e-6f);
        __bf16* dst = feat + e * FSTRIDE + 16 + gg * 16;
#pragma unroll
        for (int r = 0; r < NUM_RBF; ++r) {
          float mu = 2.0f + (float)r * (20.0f / 15.0f);
          float t = (D - mu) * 0.8f;          // / sigma(=1.25)
          dst[r] = (__bf16)__expf(-t * t);
        }
      }
      if (g == 15) {  // positional embedding row gather
        int off  = Ridx[(b << 10) + i] - Ridx[(b << 10) + j];
        int same = (chain[(b << 10) + i] == chain[(b << 10) + j]) ? 1 : 0;
        int d = off + MAX_REL;
        d = d < 0 ? 0 : (d > 2 * MAX_REL ? 2 * MAX_REL : d);
        d = same ? d : (2 * MAX_REL + 1);
#pragma unroll
        for (int f = 0; f < 16; ++f)
          feat[e * FSTRIDE + f] = (__bf16)(s_posw[d * 16 + f] + s_posb[f]);
      }
    }
    __syncthreads();

    // ---- WMMA: 16x16 tile per wave, K = 13 chunks of 32 --------------------
    v8f acc = {};
    const int n0 = wave * 16;
    const int mrow = lane & 15;
    const int hi   = (lane & 16) ? 1 : 0;
#pragma unroll
    for (int c = 0; c < KCHUNKS; ++c) {
      // A fragment: lanes 0-15 hold K {c*32+0..7, +16..23}; lanes 16-31 shift +8
      int k0a = c * 32 + (hi ? 8 : 0);
      v8bf alo = *(const v8bf*)(feat + mrow * FSTRIDE + k0a);
      v8bf ahi = *(const v8bf*)(feat + mrow * FSTRIDE + k0a + 16);
      v16bf a;
#pragma unroll
      for (int t = 0; t < 8; ++t) { a[t] = alo[t]; a[t + 8] = ahi[t]; }
      // B fragment: column = n0 + (lane&15); lanes 0-15 K {0..7, 16..23}; lanes 16-31 shift +8
      int kb0 = c * 32 + (hi ? 8 : 0);
      const __bf16* wp = wT + (n0 + mrow) * WSTRIDE + kb0;
      v8bf blo = *(const v8bf*)(wp);
      v8bf bhi = *(const v8bf*)(wp + 16);
      v16bf bv;
#pragma unroll
      for (int t = 0; t < 8; ++t) { bv[t] = blo[t]; bv[t + 8] = bhi[t]; }
      acc = __builtin_amdgcn_wmma_f32_16x16x32_bf16(
          false, a, false, bv, (short)0, acc, false, false);
    }

    // ---- epilogue: C/D layout — VGPR v -> row v (lanes<16) or v+8 ----------
    const int col = n0 + mrow;
    const float bias = s_edgeb[col];
    const int mb = hi ? 8 : 0;
#pragma unroll
    for (int v = 0; v < 8; ++v) {
      int m = mb + v;
      outE[(size_t)(tileBase + m) * EDGE_F + col] = acc[v] + bias;
    }
  }
}

// ---------------- launcher ---------------------------------------------------
extern "C" void kernel_launch(void* const* d_in, const int* in_sizes, int n_in,
                              void* d_out, int out_size, void* d_ws, size_t ws_size,
                              hipStream_t stream) {
  const float* X      = (const float*)d_in[0];
  const float* mask   = (const float*)d_in[1];
  const int*   Ridx   = (const int*)d_in[2];
  const int*   chain  = (const int*)d_in[3];
  const int*   mlbl   = (const int*)d_in[4];
  const float* pos_w  = (const float*)d_in[5];
  const float* pos_b  = (const float*)d_in[6];
  const float* edge_w = (const float*)d_in[7];
  const float* edge_b = (const float*)d_in[8];
  const float* node_w = (const float*)d_in[9];
  const float* node_b = (const float*)d_in[10];

  float* out      = (float*)d_out;
  float* outV     = out;                                      // B*N*128
  float* outE     = outV + (size_t)BB * NN * NODE_F;          // B*N*48*128
  float* outEidx  = outE + (size_t)NEDGE * EDGE_F;            // B*N*48 (as float)

  float* coords = (float*)d_ws;                               // B*N*16 f32
  int*   eidx   = (int*)((char*)d_ws + (size_t)BB * NN * 16 * 4);

  coords_kernel<<<(BB * NN + 255) / 256, 256, 0, stream>>>(X, coords, BB * NN);
  node_kernel<<<BB * NN, NODE_F, 0, stream>>>(mlbl, node_w, node_b, outV);
  topk_kernel<<<BB * NN, 256, 0, stream>>>(coords, mask, eidx, outEidx);

  size_t smem = (size_t)(EDGE_F * WSTRIDE + 16 * FSTRIDE) * sizeof(__bf16)
              + (size_t)(66 * 16 + 16 + 128) * sizeof(float);   // ~124.7 KB (<320 KB WGP LDS)
  int nblocks = NEDGE / (TILES_PER_BLOCK * 16);                 // 768
  edge_kernel<<<nblocks, 256, smem, stream>>>(coords, eidx, Ridx, chain,
                                              pos_w, pos_b, edge_w, edge_b, outE);
}
